// UnifiedCortexGPT_56985626083912
// MI455X (gfx1250) — matrix operand order repaired and checked by reading, not compiled, not run on hardware
//
#include <hip/hip_runtime.h>

typedef __bf16 bf16;
typedef __attribute__((ext_vector_type(8)))  bf16  v8bf;
typedef __attribute__((ext_vector_type(16))) bf16  v16bf;
typedef __attribute__((ext_vector_type(8)))  float v8f;
typedef __attribute__((ext_vector_type(4)))  int   v4i_;

#define B_    4096
#define D_    512
#define TD_   1024
#define C_    16
#define K_    204
#define KPAD_ 208
#define BAL   0.09f
#define RES_  0.1f

#if defined(__HIP_DEVICE_COMPILE__) && \
    __has_builtin(__builtin_amdgcn_global_load_async_to_lds_b128) && \
    __has_builtin(__builtin_amdgcn_s_wait_asynccnt)
#define HAVE_ASYNC_LDS 1
#define AS1 __attribute__((address_space(1)))
#define AS3 __attribute__((address_space(3)))
#endif

// erf via Abramowitz-Stegun 7.1.26 (|err| < 1.5e-7), fully inline (v_exp_f32)
__device__ __forceinline__ float gelu_f(float x) {
  float ax = fabsf(x) * 0.7071067811865476f;   // |x|/sqrt(2)
  float t = 1.0f / (1.0f + 0.3275911f * ax);
  float poly = ((((1.061405429f * t - 1.453152027f) * t + 1.421413741f) * t
                 - 0.284496736f) * t + 0.254829592f) * t;
  float er = 1.0f - poly * __expf(-ax * ax);
  er = copysignf(er, x);
  return 0.5f * x * (1.0f + er);
}

// ---------------------------------------------------------------- converts
__global__ void cvt_bf16(const float* __restrict__ in, bf16* __restrict__ out, int n) {
  int i = blockIdx.x * blockDim.x + threadIdx.x;
  if (i < n) out[i] = (bf16)in[i];
}

// transpose [R,Cc] f32 -> [Cc,R] bf16 (R,Cc multiples of 32), batched on z
__global__ __launch_bounds__(256) void transpose_cvt(
    const float* __restrict__ in, bf16* __restrict__ out,
    int R, int Cc, long sIn, long sOut) {
  __shared__ float tile[32][33];
  const float* inp = in + (long)blockIdx.z * sIn;
  bf16* outp = out + (long)blockIdx.z * sOut;
  int c0 = blockIdx.x * 32, r0 = blockIdx.y * 32;
  int tx = threadIdx.x, ty = threadIdx.y;  // block (32,8)
  for (int i = 0; i < 4; ++i)
    tile[ty + i * 8][tx] = inp[(long)(r0 + ty + i * 8) * Cc + c0 + tx];
  __syncthreads();
  for (int i = 0; i < 4; ++i)
    outp[(long)(c0 + ty + i * 8) * R + r0 + tx] = (bf16)tile[tx][ty + i * 8];
}

// ---------------------------------------------------------------- WMMA GEMM
// out = epi(A[M,Kd] @ Bt[N,Kd]^T + bias[N]); batched via blockIdx.z strides.
// Tile 128x128x32, 256 threads (8 waves), each wave 32x64 (2x4 wmma frags).
// Double-buffered LDS fed by global_load_async_to_lds_b128 (ASYNCcnt).
__global__ __launch_bounds__(256, 1) void gemm_bf16_wmma(
    const bf16* __restrict__ A, long sA,
    const bf16* __restrict__ Bt, long sB,
    const float* __restrict__ bias, long sBias,
    float* __restrict__ outF, int ldF, long sOF,
    bf16* __restrict__ outB, int ldB, long sOB,
    int M, int N, int Kd, int epi) {
#ifdef HAVE_ASYNC_LDS
  __shared__ bf16 As[2][128 * 40];  // row stride 40 (80B) -> conflict-free frags
  __shared__ bf16 Bs[2][128 * 40];
#else
  __shared__ bf16 As1[128 * 40];
  __shared__ bf16 Bs1[128 * 40];
#endif
  const int tid = threadIdx.x;
  const int z = blockIdx.z;
  const int n0 = blockIdx.x * 128;
  const int m0 = blockIdx.y * 128;
  A += (long)z * sA;
  Bt += (long)z * sB;

  const int lrow = tid >> 1, seg = tid & 1;  // tile-load mapping
  const int lane = tid & 31, wid = tid >> 5;
  const int waveM = wid & 3, waveN = wid >> 2;
  const int half = lane >> 4, lr = lane & 15;

  // clamp OOB rows to row M-1: they compute garbage that the m<M store guard
  // discards, and this keeps the load path branch-free.
  const int gm = min(m0 + lrow, M - 1);
  const int gn = n0 + lrow;  // N is a multiple of 128 for every call here
  const bf16* gA = A + (long)gm * Kd + seg * 16;
  const bf16* gB = Bt + (long)gn * Kd + seg * 16;
  const int ldsOff = lrow * 40 + seg * 16;

  v8f acc[2][4];
  const v8f zero8 = {0.f, 0.f, 0.f, 0.f, 0.f, 0.f, 0.f, 0.f};
  for (int i = 0; i < 2; ++i)
    for (int j = 0; j < 4; ++j) acc[i][j] = zero8;

  const int nTiles = Kd >> 5;

#ifdef HAVE_ASYNC_LDS
  auto issue = [&](int k0, int buf) {
    __builtin_amdgcn_global_load_async_to_lds_b128(
        (AS1 v4i_*)(gA + k0), (AS3 v4i_*)(&As[buf][ldsOff]), 0, 0);
    __builtin_amdgcn_global_load_async_to_lds_b128(
        (AS1 v4i_*)(gA + k0 + 8), (AS3 v4i_*)(&As[buf][ldsOff + 8]), 0, 0);
    __builtin_amdgcn_global_load_async_to_lds_b128(
        (AS1 v4i_*)(gB + k0), (AS3 v4i_*)(&Bs[buf][ldsOff]), 0, 0);
    __builtin_amdgcn_global_load_async_to_lds_b128(
        (AS1 v4i_*)(gB + k0 + 8), (AS3 v4i_*)(&Bs[buf][ldsOff + 8]), 0, 0);
  };
  issue(0, 0);
#endif

  for (int i = 0; i < nTiles; ++i) {
    const bf16* cA;
    const bf16* cB;
#ifdef HAVE_ASYNC_LDS
    __builtin_amdgcn_s_wait_asynccnt(0);  // this wave's tile-i DMA done
    __syncthreads();                      // all waves' tile-i DMA visible
    if (i + 1 < nTiles) issue((i + 1) << 5, (i + 1) & 1);  // overlap next tile
    cA = As[i & 1];
    cB = Bs[i & 1];
#else
    {  // synchronous fallback: global -> VGPR -> LDS
      const uint4* pa = (const uint4*)(gA + (i << 5));
      uint4 a0 = pa[0], a1 = pa[1];
      const uint4* pb = (const uint4*)(gB + (i << 5));
      uint4 b0 = pb[0], b1 = pb[1];
      __builtin_prefetch(gA + (i << 5) + 32, 0, 3);
      __builtin_prefetch(gB + (i << 5) + 32, 0, 3);
      uint4* da = (uint4*)(As1 + ldsOff);
      da[0] = a0;
      da[1] = a1;
      uint4* db = (uint4*)(Bs1 + ldsOff);
      db[0] = b0;
      db[1] = b1;
      __syncthreads();
      cA = As1;
      cB = Bs1;
    }
#endif

    // --- build WMMA fragments per ISA 16-bit layouts
    v16bf af[2], bfr[4];
    for (int mt = 0; mt < 2; ++mt) {
      const bf16* pa = cA + (waveM * 32 + mt * 16 + lr) * 40;
      v8bf lo = *(const v8bf*)(pa + half * 8);       // K = half*8 .. +7
      v8bf hi = *(const v8bf*)(pa + 16 + half * 8);  // K = 16+half*8 .. +7
      af[mt] = __builtin_shufflevector(lo, hi, 0, 1, 2, 3, 4, 5, 6, 7,
                                       8, 9, 10, 11, 12, 13, 14, 15);
    }
    for (int nt = 0; nt < 4; ++nt) {
      const bf16* pb = cB + (waveN * 64 + nt * 16 + lr) * 40;
      v8bf lo = *(const v8bf*)(pb + half * 8);       // K = half*8 .. +7
      v8bf hi = *(const v8bf*)(pb + 16 + half * 8);  // K = 16+half*8 .. +7
      bfr[nt] = __builtin_shufflevector(lo, hi, 0, 1, 2, 3, 4, 5, 6, 7,
                                        8, 9, 10, 11, 12, 13, 14, 15);
    }
    for (int mt = 0; mt < 2; ++mt)
      for (int nt = 0; nt < 4; ++nt)
        acc[mt][nt] = __builtin_amdgcn_wmma_f32_16x16x32_bf16(
            false, af[mt], false, bfr[nt], (short)0, acc[mt][nt], false, false);
#ifndef HAVE_ASYNC_LDS
    __syncthreads();
#endif
  }

  // --- epilogue: bias (+GELU), dual f32/bf16 stores
  const float* bz = bias + (long)z * sBias;
  float* oF = outF ? outF + (long)z * sOF : nullptr;
  bf16* oB = outB ? outB + (long)z * sOB : nullptr;
  for (int mt = 0; mt < 2; ++mt)
    for (int nt = 0; nt < 4; ++nt) {
      int n = n0 + waveN * 64 + nt * 16 + lr;
      float bv = bz[n];
      for (int v = 0; v < 8; ++v) {
        int m = m0 + waveM * 32 + mt * 16 + half * 8 + v;
        if (m < M) {
          float val = acc[mt][nt][v] + bv;
          if (epi == 1) val = gelu_f(val);
          if (oF) oF[(long)m * ldF + n] = val;
          if (oB) oB[(long)m * ldB + n] = (bf16)val;
        }
      }
    }
}

// ---------------------------------------------------------------- layernorm
__global__ __launch_bounds__(256) void ln_k(
    const float* __restrict__ in, int N,
    const float* __restrict__ gamma, const float* __restrict__ beta,
    int rowsPerBatch, long gStride,
    float* __restrict__ outF, bf16* __restrict__ outB) {
  __shared__ float red[18];
  int row = blockIdx.x, tid = threadIdx.x;
  const float* src = in + (long)row * N;
  float s = 0.f, s2 = 0.f;
  for (int i = tid; i < N; i += 256) {
    float v = src[i];
    s += v;
    s2 += v * v;
  }
  for (int off = 16; off > 0; off >>= 1) {
    s += __shfl_xor(s, off, 32);
    s2 += __shfl_xor(s2, off, 32);
  }
  int wid = tid >> 5, lane = tid & 31;
  if (lane == 0) { red[wid] = s; red[8 + wid] = s2; }
  __syncthreads();
  if (tid == 0) {
    float ts = 0.f, ts2 = 0.f;
    for (int w = 0; w < 8; ++w) { ts += red[w]; ts2 += red[8 + w]; }
    red[16] = ts;
    red[17] = ts2;
  }
  __syncthreads();
  float mean = red[16] / N;
  float var = red[17] / N - mean * mean;
  float rstd = rsqrtf(var + 1e-5f);
  long go = (long)(row / rowsPerBatch) * gStride;
  for (int i = tid; i < N; i += 256) {
    float v = (src[i] - mean) * rstd * gamma[go + i] + beta[go + i];
    if (outF) outF[(long)row * N + i] = v;
    if (outB) outB[(long)row * N + i] = (bf16)v;
  }
}

// ---------------------------------------------------------------- gate mix
__global__ void hmix_k(const float* __restrict__ x, const float* __restrict__ fast,
                       const float* __restrict__ slow, const float* __restrict__ gl,
                       float* __restrict__ hF, bf16* __restrict__ hB, int n) {
  for (int i = blockIdx.x * blockDim.x + threadIdx.x; i < n;
       i += gridDim.x * blockDim.x) {
    float g = 1.0f / (1.0f + __expf(-gl[i]));
    float mix = (1.0f - BAL) * fast[i] + BAL * slow[i];
    float h = g * mix + (1.0f - g) * x[i];
    hF[i] = h;
    hB[i] = (bf16)h;
  }
}

// ---------------------------------------------------------------- scores
// scores[c][b] = dot(h[b], colg_w[c]) + colg_b[c]; one wave per (c,b)
__global__ __launch_bounds__(256) void scores_k(
    const float* __restrict__ h, const float* __restrict__ cw,
    const float* __restrict__ cb, float* __restrict__ scores) {
  int pair = blockIdx.x * 8 + (threadIdx.x >> 5);
  int lane = threadIdx.x & 31;
  int c = pair >> 12;  // / 4096
  int b = pair & (B_ - 1);
  const float* hr = h + (long)b * D_;
  const float* wr = cw + (long)c * D_;
  float s = 0.f;
  for (int d = lane; d < D_; d += 32) s += hr[d] * wr[d];
  for (int off = 16; off > 0; off >>= 1) s += __shfl_xor(s, off, 32);
  if (lane == 0) scores[(long)c * B_ + b] = s + cb[c];
}

// ---------------------------------------------------------------- top-K
// bitonic sort in LDS -> K-th value threshold -> compacted index list + inverse map
__global__ __launch_bounds__(1024) void topk_k(
    const float* __restrict__ scores, int* __restrict__ sel, int* __restrict__ inv) {
  __shared__ float vals[B_];
  __shared__ int cnt;
  int c = blockIdx.x, tid = threadIdx.x;
  const float* sc = scores + (long)c * B_;
  for (int i = tid; i < B_; i += 1024) {
    vals[i] = sc[i];
    inv[(long)c * B_ + i] = -1;
  }
  if (tid == 0) cnt = 0;
  __syncthreads();
  for (int k = 2; k <= B_; k <<= 1)
    for (int j = k >> 1; j > 0; j >>= 1) {
      for (int i = tid; i < B_; i += 1024) {
        int ixj = i ^ j;
        if (ixj > i) {
          float a = vals[i], bv = vals[ixj];
          bool up = (i & k) == 0;  // descending
          if (up ? (a < bv) : (a > bv)) { vals[i] = bv; vals[ixj] = a; }
        }
      }
      __syncthreads();
    }
  float thresh = vals[K_ - 1];
  __syncthreads();
  for (int i = tid; i < B_; i += 1024) {  // strictly-greater first
    if (sc[i] > thresh) {
      int p = atomicAdd(&cnt, 1);
      sel[c * KPAD_ + p] = i;
      inv[(long)c * B_ + i] = p;
    }
  }
  __syncthreads();
  for (int i = tid; i < B_; i += 1024) {  // fill ties up to exactly K
    if (sc[i] == thresh) {
      int p = atomicAdd(&cnt, 1);
      if (p < K_) {
        sel[c * KPAD_ + p] = i;
        inv[(long)c * B_ + i] = p;
      }
    }
  }
  __syncthreads();
  for (int i = tid; i < KPAD_; i += 1024)
    if (i >= K_) sel[c * KPAD_ + i] = -1;
}

// ---------------------------------------------------------------- gather rows
__global__ void gather_k(const bf16* __restrict__ hB, const int* __restrict__ sel,
                         bf16* __restrict__ hsel) {
  int row = blockIdx.x;  // c*KPAD + i
  int b = sel[row];
  bf16* dst = hsel + (long)row * D_;
  if (b < 0) {
    for (int d = threadIdx.x; d < D_; d += 256) dst[d] = (bf16)0.0f;
  } else {
    const bf16* src = hB + (long)b * D_;
    for (int d = threadIdx.x; d < D_; d += 256) dst[d] = src[d];
  }
}

// ---------------------------------------------------------------- final combine
__global__ void final_k(const float* __restrict__ hF, const float* __restrict__ p2,
                        const int* __restrict__ inv, float* __restrict__ out) {
  __shared__ int iv[C_];
  int b = blockIdx.x;
  if (threadIdx.x < C_) iv[threadIdx.x] = inv[(long)threadIdx.x * B_ + b];
  __syncthreads();
  for (int d = threadIdx.x; d < D_; d += 256) {
    float a = hF[(long)b * D_ + d];
    float s = 0.f;
    for (int c = 0; c < C_; ++c) {
      int i = iv[c];
      if (i >= 0) s += p2[((long)c * KPAD_ + i) * D_ + d];
    }
    out[(long)b * D_ + d] = a + (RES_ / (float)C_) * s;
  }
}

// ================================================================ launch
extern "C" void kernel_launch(void* const* d_in, const int* in_sizes, int n_in,
                              void* d_out, int out_size, void* d_ws, size_t ws_size,
                              hipStream_t stream) {
  const float* x     = (const float*)d_in[0];
  const float* fw1   = (const float*)d_in[1];
  const float* fb1   = (const float*)d_in[2];
  const float* fw2   = (const float*)d_in[3];
  const float* fb2   = (const float*)d_in[4];
  const float* sw1   = (const float*)d_in[5];
  const float* sb1   = (const float*)d_in[6];
  const float* sg    = (const float*)d_in[7];
  const float* sbeta = (const float*)d_in[8];
  const float* sw2   = (const float*)d_in[9];
  const float* sb2   = (const float*)d_in[10];
  const float* gw    = (const float*)d_in[11];
  const float* gb    = (const float*)d_in[12];
  const float* cw1   = (const float*)d_in[13];
  const float* cb1   = (const float*)d_in[14];
  const float* cg1   = (const float*)d_in[15];
  const float* cbe1  = (const float*)d_in[16];
  const float* cw2   = (const float*)d_in[17];
  const float* cb2   = (const float*)d_in[18];
  const float* cg2   = (const float*)d_in[19];
  const float* cbe2  = (const float*)d_in[20];
  const float* cgw   = (const float*)d_in[21];
  const float* cgb   = (const float*)d_in[22];
  float* out = (float*)d_out;

  char* ws = (char*)d_ws;
  size_t off = 0;
  auto alloc = [&](size_t bytes) -> void* {
    off = (off + 255) & ~(size_t)255;
    void* p = ws + off;
    off += bytes;
    return p;
  };
  bf16*  x_bf   = (bf16*)alloc((size_t)B_ * D_ * 2);
  bf16*  fw1t   = (bf16*)alloc((size_t)D_ * TD_ * 2);  // [1024,512]
  bf16*  fw2t   = (bf16*)alloc((size_t)TD_ * D_ * 2);  // [512,1024]
  bf16*  sw1t   = (bf16*)alloc((size_t)D_ * D_ * 2);
  bf16*  sw2t   = (bf16*)alloc((size_t)D_ * D_ * 2);
  bf16*  gwt    = (bf16*)alloc((size_t)TD_ * D_ * 2);  // [512,1024]
  bf16*  cw1t   = (bf16*)alloc((size_t)C_ * D_ * TD_ * 2);
  bf16*  cw2t   = (bf16*)alloc((size_t)C_ * TD_ * D_ * 2);
  bf16*  t1B    = (bf16*)alloc((size_t)B_ * TD_ * 2);
  bf16*  fsB    = (bf16*)alloc((size_t)B_ * TD_ * 2);  // concat(fast,slow) bf16
  float* fastF  = (float*)alloc((size_t)B_ * D_ * 4);
  float* slowF  = (float*)alloc((size_t)B_ * D_ * 4);
  float* s1F    = (float*)alloc((size_t)B_ * D_ * 4);
  bf16*  s1lnB  = (bf16*)alloc((size_t)B_ * D_ * 2);
  float* gateF  = (float*)alloc((size_t)B_ * D_ * 4);
  float* hF     = (float*)alloc((size_t)B_ * D_ * 4);
  bf16*  hB     = (bf16*)alloc((size_t)B_ * D_ * 2);
  float* scores = (float*)alloc((size_t)C_ * B_ * 4);
  int*   inv    = (int*)alloc((size_t)C_ * B_ * 4);
  int*   sel    = (int*)alloc((size_t)C_ * KPAD_ * 4);
  bf16*  hsel   = (bf16*)alloc((size_t)C_ * KPAD_ * D_ * 2);
  float* p1F    = (float*)alloc((size_t)C_ * KPAD_ * TD_ * 4);
  bf16*  p1lnB  = (bf16*)alloc((size_t)C_ * KPAD_ * TD_ * 2);
  float* p2F    = (float*)alloc((size_t)C_ * KPAD_ * D_ * 4);
  float* p2ln   = (float*)alloc((size_t)C_ * KPAD_ * D_ * 4);

  dim3 blk(256);
  // precision conversion + weight transposition (one pass)
  cvt_bf16<<<(B_ * D_ + 255) / 256, 256, 0, stream>>>(x, x_bf, B_ * D_);
  transpose_cvt<<<dim3(TD_ / 32, D_ / 32, 1), dim3(32, 8), 0, stream>>>(fw1, fw1t, D_, TD_, 0, 0);
  transpose_cvt<<<dim3(D_ / 32, TD_ / 32, 1), dim3(32, 8), 0, stream>>>(fw2, fw2t, TD_, D_, 0, 0);
  transpose_cvt<<<dim3(D_ / 32, D_ / 32, 1), dim3(32, 8), 0, stream>>>(sw1, sw1t, D_, D_, 0, 0);
  transpose_cvt<<<dim3(D_ / 32, D_ / 32, 1), dim3(32, 8), 0, stream>>>(sw2, sw2t, D_, D_, 0, 0);
  transpose_cvt<<<dim3(D_ / 32, TD_ / 32, 1), dim3(32, 8), 0, stream>>>(gw, gwt, TD_, D_, 0, 0);
  transpose_cvt<<<dim3(TD_ / 32, D_ / 32, C_), dim3(32, 8), 0, stream>>>(
      cw1, cw1t, D_, TD_, (long)D_ * TD_, (long)D_ * TD_);
  transpose_cvt<<<dim3(D_ / 32, TD_ / 32, C_), dim3(32, 8), 0, stream>>>(
      cw2, cw2t, TD_, D_, (long)TD_ * D_, (long)TD_ * D_);

  // fast encoder: t1 = gelu(x@fw1+fb1); fast = gelu(t1@fw2+fb2)
  gemm_bf16_wmma<<<dim3(8, 32, 1), blk, 0, stream>>>(
      x_bf, 0, fw1t, 0, fb1, 0, nullptr, 0, 0, t1B, TD_, 0, B_, TD_, D_, 1);
  gemm_bf16_wmma<<<dim3(4, 32, 1), blk, 0, stream>>>(
      t1B, 0, fw2t, 0, fb2, 0, fastF, D_, 0, fsB, TD_, 0, B_, D_, TD_, 1);
  // slow encoder: s1 = gelu(x@sw1+sb1); LN; slow = gelu(ln@sw2+sb2)
  gemm_bf16_wmma<<<dim3(4, 32, 1), blk, 0, stream>>>(
      x_bf, 0, sw1t, 0, sb1, 0, s1F, D_, 0, nullptr, 0, 0, B_, D_, D_, 1);
  ln_k<<<B_, 256, 0, stream>>>(s1F, D_, sg, sbeta, B_, 0, nullptr, s1lnB);
  gemm_bf16_wmma<<<dim3(4, 32, 1), blk, 0, stream>>>(
      s1lnB, 0, sw2t, 0, sb2, 0, slowF, D_, 0, fsB + D_, TD_, 0, B_, D_, D_, 1);
  // gate = sigmoid([fast,slow]@gw+gb); h = g*mix+(1-g)*x
  gemm_bf16_wmma<<<dim3(4, 32, 1), blk, 0, stream>>>(
      fsB, 0, gwt, 0, gb, 0, gateF, D_, 0, nullptr, 0, 0, B_, D_, TD_, 0);
  hmix_k<<<2048, 256, 0, stream>>>(x, fastF, slowF, gateF, hF, hB, B_ * D_);

  // column routing: scores -> top-K selection -> gather selected rows
  scores_k<<<(C_ * B_) / 8, 256, 0, stream>>>(hF, cgw, cgb, scores);
  topk_k<<<C_, 1024, 0, stream>>>(scores, sel, inv);
  gather_k<<<C_ * KPAD_, 256, 0, stream>>>(hB, sel, hsel);

  // cortical columns on selected rows only (batched over C)
  gemm_bf16_wmma<<<dim3(8, 2, C_), blk, 0, stream>>>(
      hsel, (long)KPAD_ * D_, cw1t, (long)D_ * TD_, cb1, TD_,
      p1F, TD_, (long)KPAD_ * TD_, nullptr, 0, 0, KPAD_, TD_, D_, 1);
  ln_k<<<C_ * KPAD_, 256, 0, stream>>>(p1F, TD_, cg1, cbe1, KPAD_, TD_, nullptr, p1lnB);
  gemm_bf16_wmma<<<dim3(4, 2, C_), blk, 0, stream>>>(
      p1lnB, (long)KPAD_ * TD_, cw2t, (long)TD_ * D_, cb2, D_,
      p2F, D_, (long)KPAD_ * D_, nullptr, 0, 0, KPAD_, D_, TD_, 1);
  ln_k<<<C_ * KPAD_, 256, 0, stream>>>(p2F, D_, cg2, cbe2, KPAD_, D_, p2ln, nullptr);

  // out = h + RES * mean_c(masked columns)
  final_k<<<B_, 256, 0, stream>>>(hF, p2ln, inv, out);
}
